// EquivariantConvolution_43439299232024
// MI455X (gfx1250) — matrix operand order repaired and checked by reading, not compiled, not run on hardware
//
#include <hip/hip_runtime.h>
#include <cmath>
#include <complex>
#include <algorithm>

#define THREADS 128
#define TE      16
#define N_EDGES 20000
#define N_NODES 2000
#define IN_DIM  288
#define SH_DIM  9
#define RDIM    64
#define NINS    11
#define WNUM    11264
#define W3J_TOT 363

typedef _Float16 v16h __attribute__((ext_vector_type(16)));
typedef _Float16 v8h  __attribute__((ext_vector_type(8)));
typedef float    v8f  __attribute__((ext_vector_type(8)));

// instruction metadata (e3nn FullyConnectedTensorProduct enumeration)
constexpr int INS_I1[NINS] = {0,0,0,1,1,1,1,2,2,2,2};
constexpr int INS_I2[NINS] = {0,1,2,0,1,1,2,0,1,2,2};
constexpr int INS_I3[NINS] = {0,1,2,1,0,2,1,2,1,0,2};
// flattened w3j table offsets (sizes 1,9,25,9,9,45,45,25,45,25,125)
constexpr int W3OFF[NINS]  = {0,1,10,35,44,53,98,143,168,213,238};
constexpr int XOFF[3]  = {0,32,128};
constexpr int SHOFF[3] = {0,1,4};

// ---------------------------------------------------------------------------
// Kernel 1: transpose + fp16-convert W2/sqrt(64) -> W2T[n][k], k contiguous
// ---------------------------------------------------------------------------
__global__ void w2t_kernel(const float* __restrict__ W2, _Float16* __restrict__ w2t) {
  int i = blockIdx.x * blockDim.x + threadIdx.x;
  if (i >= WNUM * RDIM) return;
  int k = i & 63, n = i >> 6;
  w2t[(size_t)n * RDIM + k] = (_Float16)(W2[(size_t)k * WNUM + n] * 0.125f);
}

// ---------------------------------------------------------------------------
// Kernel 2: self-connection  out[n, off_l + w*d + k] = (x_l @ Wsc_l)/sqrt(32)
// (writes the full output; message kernel atomically accumulates on top)
// ---------------------------------------------------------------------------
__global__ void __launch_bounds__(IN_DIM)
selfconn_kernel(const float* __restrict__ nodef,
                const float* __restrict__ w0,
                const float* __restrict__ w1,
                const float* __restrict__ w2,
                float* __restrict__ out) {
  int n = blockIdx.x, t = threadIdx.x;
  int w, k, d, xo;
  const float* W;
  if (t < 32)        { w = t;             k = 0;            d = 1; xo = 0;   W = w0; }
  else if (t < 128)  { int r = t - 32;  w = r / 3; k = r % 3; d = 3; xo = 32;  W = w1; }
  else               { int r = t - 128; w = r / 5; k = r % 5; d = 5; xo = 128; W = w2; }
  const float* xp = nodef + (size_t)n * IN_DIM + xo + k;
  float a = 0.f;
  #pragma unroll 8
  for (int u = 0; u < 32; ++u) a += xp[u * d] * W[u * 32 + w];
  out[(size_t)n * IN_DIM + t] = a * 0.17677669529663689f; // 1/sqrt(32)
}

// ---------------------------------------------------------------------------
// apply phase: out[e,w,k] += sum_{u in chunk} wgen[e,u,w] * z[e,u,k]
// Template on (D3, OB) so the accumulator is statically indexed -> stays in
// VGPRs (no scratch spills).
// ---------------------------------------------------------------------------
template<int D3, int OB>
__device__ __forceinline__ void apply_chunk(float (&acc)[4][9],
                                            const float* __restrict__ z_s,
                                            const float* __restrict__ wg_s,
                                            int tid, int uc) {
  #pragma unroll
  for (int p = 0; p < 4; ++p) {
    int i = p * THREADS + tid;
    int e = i >> 5, w = i & 31;
    const float* zb = z_s + e * 160;          // z[e][u][k], 32*5 per edge
    const float* wb = wg_s + e * 256 + w;     // wgen chunk [e][ul*32 + w]
    #pragma unroll
    for (int ul = 0; ul < 8; ++ul) {
      float wv = wb[ul * 32];
      const float* zp = zb + (uc * 8 + ul) * 5;
      #pragma unroll
      for (int k = 0; k < D3; ++k)
        acc[p][OB + k] += wv * zp[k];
    }
  }
}

// ---------------------------------------------------------------------------
// Kernel 3: fused  h -> (WMMA weight-gen) -> tensor product -> scatter-add
// One workgroup = 16 edges, 128 threads = 4 waves.
// ---------------------------------------------------------------------------
__global__ void __launch_bounds__(THREADS)
tp_fused_kernel(const float* __restrict__ nodef,
                const int*   __restrict__ eidx,
                const float* __restrict__ esh,
                const float* __restrict__ erad,
                const float* __restrict__ W1,
                const _Float16* __restrict__ w2t,
                const float* __restrict__ w3j_g,
                float* __restrict__ out) {
  __shared__ float    x_s[TE][IN_DIM];     // gathered node features
  __shared__ float    sh_s[TE][12];        // spherical harmonics (padded)
  __shared__ _Float16 h_s[TE][RDIM];       // radial MLP hidden (fp16, WMMA A)
  __shared__ float    z_s[TE * 32 * 5];    // z[e][u][k]
  __shared__ float    wg_s[TE * 256];      // generated-weight chunk (u-block of 8)
  __shared__ float    w3j_s[W3J_TOT];

  const int tid   = threadIdx.x;
  const int ebase = blockIdx.x * TE;

  for (int i = tid; i < W3J_TOT; i += THREADS) w3j_s[i] = w3j_g[i];
  for (int i = tid; i < TE * IN_DIM; i += THREADS) {
    int e = i / IN_DIM, c = i % IN_DIM;
    int src = eidx[ebase + e];                       // edge_index row 0
    x_s[e][c] = nodef[(size_t)src * IN_DIM + c];
  }
  for (int i = tid; i < TE * SH_DIM; i += THREADS) {
    int e = i / SH_DIM, c = i % SH_DIM;
    sh_s[e][c] = esh[(size_t)(ebase + e) * SH_DIM + c];
  }
  // h = 1.679 * silu(edge_radial @ W1 / 8), stored fp16
  for (int p = 0; p < (TE * RDIM) / THREADS; ++p) {
    int i = p * THREADS + tid;
    int e = i >> 6, c = i & 63;
    const float* rp = erad + (size_t)(ebase + e) * RDIM;
    float a = 0.f;
    #pragma unroll 8
    for (int r = 0; r < RDIM; ++r) a += rp[r] * W1[r * RDIM + c];
    a *= 0.125f;
    float s = a / (1.f + __expf(-a));
    h_s[e][c] = (_Float16)(1.6790f * s);
  }
  __syncthreads();

  const int lane = tid & 31, wave = tid >> 5;
  const int m = lane & 15, half = lane >> 4, col = lane & 15;

  // A fragments (edges x h-features), reused for all 11 instructions.
  // 16-bit A layout: lane holds row m; elems k = half*16+0..15 (contiguous, matches B).
  v16h a0, a1;
  ((v8h*)&a0)[0] = *(const v8h*)&h_s[m][half * 16];
  ((v8h*)&a0)[1] = *(const v8h*)&h_s[m][half * 16 + 8];
  ((v8h*)&a1)[0] = *(const v8h*)&h_s[m][32 + half * 16];
  ((v8h*)&a1)[1] = *(const v8h*)&h_s[m][40 + half * 16];

  // per-thread accumulators: 4 (edge,w) pairs x 9 output slots (1+3+5)
  float acc[4][9];
  #pragma unroll
  for (int p = 0; p < 4; ++p)
    #pragma unroll
    for (int s = 0; s < 9; ++s) acc[p][s] = 0.f;

  for (int t = 0; t < NINS; ++t) {
    const int i1 = INS_I1[t], i2 = INS_I2[t], i3 = INS_I3[t];
    const int d1 = 2 * i1 + 1, d2 = 2 * i2 + 1, d3 = 2 * i3 + 1;
    const int xo = XOFF[i1], so = SHOFF[i2], wo = W3OFF[t];

    // z[e][u][k] = sum_{i,j} W3J[i,j,k] * x[e,u,i] * sh[e,j]
    for (int p = 0; p < 4; ++p) {
      int i = p * THREADS + tid;
      int e = i >> 5, u = i & 31;
      const float* xp = &x_s[e][xo + u * d1];
      const float* sp = &sh_s[e][so];
      float* zp = &z_s[(e * 32 + u) * 5];
      for (int k = 0; k < d3; ++k) {
        float a = 0.f;
        for (int ii = 0; ii < d1; ++ii) {
          float xv = xp[ii];
          for (int j = 0; j < d2; ++j)
            a += w3j_s[wo + (ii * d2 + j) * d3 + k] * xv * sp[j];
        }
        zp[k] = a;
      }
    }

    // 4 u-chunks of 8: per chunk generate w-tile [16 edges x 256] via WMMA
    for (int uc = 0; uc < 4; ++uc) {
      // uniform, fully unrolled tile loop -> straight-line WMMA, EXEC all-ones
      #pragma unroll
      for (int it = 0; it < 4; ++it) {
        int nt = it * 4 + wave;
        int ncol = t * 1024 + uc * 256 + nt * 16 + col;     // global (u,w) column
        const _Float16* bp = w2t + (size_t)ncol * RDIM + half * 16;
        v16h b0 = *(const v16h*)bp;          // k =  0..31 slice for this lane
        v16h b1 = *(const v16h*)(bp + 32);   // k = 32..63 slice
        v8f c = {};
        c = __builtin_amdgcn_wmma_f32_16x16x32_f16(false, a0, false, b0,
                                                   (short)0, c, false, false);
        c = __builtin_amdgcn_wmma_f32_16x16x32_f16(false, a1, false, b1,
                                                   (short)0, c, false, false);
        #pragma unroll
        for (int r = 0; r < 8; ++r)          // D row = r + half*8 (C/D layout)
          wg_s[(r + half * 8) * 256 + nt * 16 + col] = c[r];
      }
      __syncthreads();

      if (i3 == 0)      apply_chunk<1, 0>(acc, z_s, wg_s, tid, uc);
      else if (i3 == 1) apply_chunk<3, 1>(acc, z_s, wg_s, tid, uc);
      else              apply_chunk<5, 4>(acc, z_s, wg_s, tid, uc);
      __syncthreads();
    }
  }

  // scale + scatter-add messages into destination nodes
  const float PW0 = 0.10206207261596575f;  // sqrt(1/96)
  const float PW1 = 0.15309310892394863f;  // sqrt(3/128)
  const float PW2 = 0.19764235376052372f;  // sqrt(5/128)
  const float S   = 0.31622776601683794f;  // 1/sqrt(10)
  #pragma unroll
  for (int p = 0; p < 4; ++p) {
    int i = p * THREADS + tid;
    int e = i >> 5, w = i & 31;
    int dst = eidx[N_EDGES + ebase + e];               // edge_index row 1
    float* op = out + (size_t)dst * IN_DIM;
    atomicAdd(op + w, S * PW0 * acc[p][0]);
    #pragma unroll
    for (int k = 0; k < 3; ++k) atomicAdd(op + 32 + w * 3 + k, S * PW1 * acc[p][1 + k]);
    #pragma unroll
    for (int k = 0; k < 5; ++k) atomicAdd(op + 128 + w * 5 + k, S * PW2 * acc[p][4 + k]);
  }
}

// ---------------------------------------------------------------------------
// Host: Wigner-3j tables (same math as reference, deterministic each call)
// ---------------------------------------------------------------------------
static double factd(int n) { double r = 1; for (int i = 2; i <= n; ++i) r *= i; return r; }

static void qmat(int l, std::complex<double> Q[5][5]) {
  for (int a = 0; a < 5; ++a) for (int b = 0; b < 5; ++b) Q[a][b] = 0.0;
  const std::complex<double> I(0, 1);
  const double is2 = 1.0 / std::sqrt(2.0);
  for (int m = -l; m < 0; ++m) { Q[l + m][l - m] = is2; Q[l + m][l + m] = -I * is2; }
  Q[l][l] = 1.0;
  for (int m = 1; m <= l; ++m) {
    double sg = (m & 1) ? -1.0 : 1.0;
    Q[l + m][l + m] = sg * is2;
    Q[l + m][l - m] = I * sg * is2;
  }
  std::complex<double> ph(1, 0);
  for (int i = 0; i < l; ++i) ph *= std::complex<double>(0, -1);
  for (int a = 0; a < 2 * l + 1; ++a)
    for (int b = 0; b < 2 * l + 1; ++b) Q[a][b] *= ph;
}

static void fill_w3j(float* o) {
  const int ins[NINS][3] = {{0,0,0},{0,1,1},{0,2,2},{1,0,1},{1,1,0},{1,1,2},
                            {1,2,1},{2,0,2},{2,1,1},{2,2,0},{2,2,2}};
  int off = 0;
  for (int t = 0; t < NINS; ++t) {
    int j1 = ins[t][0], j2 = ins[t][1], j3 = ins[t][2];
    int d1 = 2*j1+1, d2 = 2*j2+1, d3 = 2*j3+1;
    double C[5][5][5] = {};
    for (int m1 = -j1; m1 <= j1; ++m1)
      for (int m2 = -j2; m2 <= j2; ++m2) {
        int m3 = m1 + m2;
        int am3 = m3 < 0 ? -m3 : m3;
        if (am3 > j3) continue;
        int vmin = std::max(std::max(-j1 + j2 + m3, -j1 + m1), 0);
        int vmax = std::min(std::min(j2 + j3 + m1, j3 - j1 + j2), j3 + m3);
        double c = std::sqrt((2*j3+1) * factd(j3+j1-j2) * factd(j3-j1+j2) * factd(j1+j2-j3)
                   * factd(j3+m3) * factd(j3-m3)
                   / (factd(j1+j2+j3+1) * factd(j1-m1) * factd(j1+m1)
                      * factd(j2-m2) * factd(j2+m2)));
        double s = 0;
        for (int v = vmin; v <= vmax; ++v) {
          double sg = ((v + j2 + m2) & 1) ? -1.0 : 1.0;
          s += sg * factd(j2+j3+m1-v) * factd(j1-m1+v)
               / (factd(v) * factd(j3-j1+j2-v) * factd(j3+m3-v) * factd(v+j1-j2-m3));
        }
        C[m1+j1][m2+j2][m3+j3] = c * s;
      }
    std::complex<double> Q1[5][5], Q2[5][5], Q3[5][5];
    qmat(j1, Q1); qmat(j2, Q2); qmat(j3, Q3);
    double R[5][5][5] = {};
    double nrm = 0;
    for (int a = 0; a < d1; ++a)
      for (int b = 0; b < d2; ++b)
        for (int c2 = 0; c2 < d3; ++c2) {
          std::complex<double> acc(0, 0);
          for (int i = 0; i < d1; ++i)
            for (int k = 0; k < d2; ++k)
              for (int n = 0; n < d3; ++n)
                acc += Q1[i][a] * Q2[k][b] * std::conj(Q3[n][c2]) * C[i][k][n];
          R[a][b][c2] = acc.real();
          nrm += acc.real() * acc.real();
        }
    nrm = std::sqrt(nrm);
    for (int a = 0; a < d1; ++a)
      for (int b = 0; b < d2; ++b)
        for (int c2 = 0; c2 < d3; ++c2)
          o[off + (a * d2 + b) * d3 + c2] = (float)(R[a][b][c2] / nrm);
    off += d1 * d2 * d3;
  }
}

// ---------------------------------------------------------------------------
extern "C" void kernel_launch(void* const* d_in, const int* in_sizes, int n_in,
                              void* d_out, int out_size, void* d_ws, size_t ws_size,
                              hipStream_t stream) {
  (void)in_sizes; (void)n_in; (void)out_size; (void)ws_size;

  const float* nodef = (const float*)d_in[0];
  const int*   eidx  = (const int*)  d_in[1];
  const float* esh   = (const float*)d_in[2];
  const float* erad  = (const float*)d_in[3];
  const float* W1    = (const float*)d_in[4];
  const float* W2    = (const float*)d_in[5];
  const float* Wsc0  = (const float*)d_in[6];
  const float* Wsc1  = (const float*)d_in[7];
  const float* Wsc2  = (const float*)d_in[8];
  float* out = (float*)d_out;

  // scratch layout: [ W2T fp16 : 11264*64*2 B ][ w3j : 363 floats ]
  char* ws = (char*)d_ws;
  _Float16* w2t = (_Float16*)ws;
  float* w3j_d = (float*)(ws + (size_t)WNUM * RDIM * sizeof(_Float16));

  // host-side Wigner-3j (deterministic; persistent buffer for graph replay)
  static float w3j_h[W3J_TOT];
  fill_w3j(w3j_h);
  hipMemcpyAsync(w3j_d, w3j_h, sizeof(w3j_h), hipMemcpyHostToDevice, stream);

  w2t_kernel<<<(WNUM * RDIM + 255) / 256, 256, 0, stream>>>(W2, w2t);
  selfconn_kernel<<<N_NODES, IN_DIM, 0, stream>>>(nodef, Wsc0, Wsc1, Wsc2, out);
  tp_fused_kernel<<<N_EDGES / TE, THREADS, 0, stream>>>(
      nodef, eidx, esh, erad, W1, w2t, w3j_d, out);
}
